// DynamicLSTMCell_67954972557602
// MI455X (gfx1250) — matrix operand order repaired and checked by reading, not compiled, not run on hardware
//
#include <hip/hip_runtime.h>
#include <hip/hip_bf16.h>

typedef _Float16 half8   __attribute__((ext_vector_type(8)));
typedef _Float16 half16  __attribute__((ext_vector_type(16)));
typedef float    floatx8 __attribute__((ext_vector_type(8)));

#define B_TOT   16384
#define IN_DIM  256
#define OUT_DIM 256
#define D_DIM   512            // IN + OUT
#define NCELLS  4
#define NCOLS   4096           // 4 gates * OUT * NCELLS
#define BM      32             // batch rows per workgroup (2 M-tiles)

union H16 { half16 v; half8 h[2]; };

__device__ __forceinline__ float fast_sigmoid(float x) {
    return 1.0f / (1.0f + __expf(-x));
}
__device__ __forceinline__ float fast_tanh(float x) {
    // 1 - 2/(e^{2x}+1): saturates cleanly to +/-1, no inf/inf NaN
    float t = __expf(2.0f * x);
    return 1.0f - 2.0f / (t + 1.0f);
}

// ---------------------------------------------------------------------------
// Kernel 1: transpose W_gates (512 x 4096, row-major [k][n]) into
// column-major-per-output f16 hi/lo arrays: Whi/Wlo[n*512 + k].
// LDS-tiled 32x32 so both the global read and the global write coalesce.
// ---------------------------------------------------------------------------
__global__ __launch_bounds__(256) void transpose_convert(
    const float* __restrict__ W, _Float16* __restrict__ Whi,
    _Float16* __restrict__ Wlo)
{
    __shared__ float tile[32][33];
    const int bn = blockIdx.x * 32;   // n block: 4096/32 = 128
    const int bk = blockIdx.y * 32;   // k block:  512/32 = 16
    const int t  = threadIdx.x;

    for (int i = t; i < 1024; i += 256) {
        int kk = i >> 5, nn = i & 31;                    // consecutive t -> consecutive n
        tile[kk][nn] = W[(size_t)(bk + kk) * NCOLS + bn + nn];
    }
    __syncthreads();
    for (int i = t; i < 1024; i += 256) {
        int nn = i >> 5, kk = i & 31;                    // consecutive t -> consecutive k
        float v = tile[kk][nn];
        _Float16 hi = (_Float16)v;
        size_t idx = (size_t)(bn + nn) * D_DIM + bk + kk;
        Whi[idx] = hi;
        Wlo[idx] = (_Float16)(v - (float)hi);
    }
}

// ---------------------------------------------------------------------------
// Kernel 2: fused MoE-LSTM cell.
// One workgroup = 32 batch rows (2 M-tiles), 512 threads = 16 waves.
// Wave w owns output columns [w*16, w*16+16) for all 4 gates of each cell
// => per wave: 4 gate column-tiles x 2 M-tiles = 8 WMMA accumulator tiles.
// Each B fragment is reused across both M-tiles (6 WMMAs per B-tile load).
// Split-precision GEMM: acc += Ahi*Bhi + Ahi*Blo + Alo*Bhi (f32 accumulate).
// ---------------------------------------------------------------------------
__global__ __launch_bounds__(512) void lstm_moe_kernel(
    const float* __restrict__ x, const float* __restrict__ c_in,
    const float* __restrict__ h, const float* __restrict__ b_gates,
    const float* __restrict__ W_ctrl, const float* __restrict__ b_ctrl,
    const _Float16* __restrict__ Whi, const _Float16* __restrict__ Wlo,
    float* __restrict__ out)
{
    __shared__ _Float16 Ahi[BM * D_DIM];   // 32 KB
    __shared__ _Float16 Alo[BM * D_DIM];   // 32 KB
    __shared__ float logits_s[BM][4];
    __shared__ float gate_s[BM][4];

    const int tid  = threadIdx.x;
    const int row0 = blockIdx.x * BM;

    // ---- Stage feats = [x | h] into LDS as hi/lo f16 (coalesced over k) ----
    for (int i = tid; i < BM * D_DIM; i += 512) {
        int r = i >> 9;
        int k = i & (D_DIM - 1);
        float v = (k < IN_DIM) ? x[(size_t)(row0 + r) * IN_DIM + k]
                               : h[(size_t)(row0 + r) * OUT_DIM + (k - IN_DIM)];
        _Float16 hi = (_Float16)v;
        Ahi[i] = hi;
        Alo[i] = (_Float16)(v - (float)hi);
    }

    // ---- Control logits: 32 rows x 4 cells, K = 512 (tiny, plain VALU) ----
    if (tid < 4 * BM) {
        int r = tid >> 2, n = tid & 3;
        const float* xr = x + (size_t)(row0 + r) * IN_DIM;
        const float* hr = h + (size_t)(row0 + r) * OUT_DIM;
        float acc = b_ctrl[n];
        for (int k = 0; k < IN_DIM; ++k)  acc += xr[k] * W_ctrl[k * 4 + n];
        for (int k = 0; k < OUT_DIM; ++k) acc += hr[k] * W_ctrl[(IN_DIM + k) * 4 + n];
        logits_s[r][n] = acc;
    }
    __syncthreads();

    // ---- Top-2 softmax gate per row (strict '>' matches top_k tie-break) ----
    if (tid < BM) {
        float l[4];
        #pragma unroll
        for (int n = 0; n < 4; ++n) l[n] = logits_s[tid][n];
        int i1 = 0;
        #pragma unroll
        for (int n = 1; n < 4; ++n) if (l[n] > l[i1]) i1 = n;
        int i2 = -1;
        #pragma unroll
        for (int n = 0; n < 4; ++n) {
            if (n == i1) continue;
            if (i2 < 0 || l[n] > l[i2]) i2 = n;
        }
        float e2 = __expf(l[i2] - l[i1]);   // e1 == 1
        float s  = 1.0f + e2;
        #pragma unroll
        for (int n = 0; n < 4; ++n) gate_s[tid][n] = 0.0f;
        gate_s[tid][i1] = 1.0f / s;
        gate_s[tid][i2] = e2 / s;
    }
    __syncthreads();

    const int wave   = tid >> 5;      // 0..15
    const int lane   = tid & 31;
    const int laneHi = lane >> 4;     // half-wave select
    const int lmod   = lane & 15;
    const int o      = wave * 16 + lmod;   // this lane's output column

    // c-input fragments + persistent nh/nc accumulators, per M-tile.
    // D-matrix layout: local row = m*16 + j + 8*laneHi, col = lmod.
    floatx8 cfrag[2], nh_acc[2], nc_acc[2];
    #pragma unroll
    for (int m = 0; m < 2; ++m) {
        #pragma unroll
        for (int j = 0; j < 8; ++j) {
            const int lr = m * 16 + j + 8 * laneHi;
            cfrag[m][j]  = c_in[(size_t)(row0 + lr) * OUT_DIM + o];
            nh_acc[m][j] = 0.0f;
            nc_acc[m][j] = 0.0f;
        }
    }

    #pragma unroll 1
    for (int cell = 0; cell < NCELLS; ++cell) {
        floatx8 acc[4][2];   // [gate][m-tile]
        #pragma unroll
        for (int g = 0; g < 4; ++g)
            #pragma unroll
            for (int m = 0; m < 2; ++m)
                #pragma unroll
                for (int j = 0; j < 8; ++j) acc[g][m][j] = 0.0f;

        #pragma unroll 1
        for (int kk = 0; kk < D_DIM; kk += 32) {
            // A fragments (16x32 f16), one per M-tile:
            // lane<16: halves0-7=K0-7, 8-15=K16-23; lane>=16: 8-15 / 24-31.
            H16 a_hi[2], a_lo[2];
            const int kA0 = kk + laneHi * 8;
            const int kA1 = kk + 16 + laneHi * 8;
            #pragma unroll
            for (int m = 0; m < 2; ++m) {
                const int rowOff = (m * 16 + lmod) * D_DIM;
                a_hi[m].h[0] = *(const half8*)&Ahi[rowOff + kA0];
                a_hi[m].h[1] = *(const half8*)&Ahi[rowOff + kA1];
                a_lo[m].h[0] = *(const half8*)&Alo[rowOff + kA0];
                a_lo[m].h[1] = *(const half8*)&Alo[rowOff + kA1];
            }
            const int kB = kk + laneHi * 8;    // B: K 8h..8h+7 and 16+8h..16+8h+7 per half-wave

            #pragma unroll
            for (int g = 0; g < 4; ++g) {
                const int n = cell * 1024 + g * 256 + o;
                const _Float16* bp_hi = Whi + (size_t)n * D_DIM + kB;
                const _Float16* bp_lo = Wlo + (size_t)n * D_DIM + kB;
                H16 b_hi, b_lo;
                b_hi.h[0] = *(const half8*)(bp_hi);
                b_hi.h[1] = *(const half8*)(bp_hi + 16);
                b_lo.h[0] = *(const half8*)(bp_lo);
                b_lo.h[1] = *(const half8*)(bp_lo + 16);
                #pragma unroll
                for (int m = 0; m < 2; ++m) {
                    floatx8 d = acc[g][m];
                    d = __builtin_amdgcn_wmma_f32_16x16x32_f16(
                            false, a_hi[m].v, false, b_hi.v, (short)0, d, false, false);
                    d = __builtin_amdgcn_wmma_f32_16x16x32_f16(
                            false, a_hi[m].v, false, b_lo.v, (short)0, d, false, false);
                    d = __builtin_amdgcn_wmma_f32_16x16x32_f16(
                            false, a_lo[m].v, false, b_hi.v, (short)0, d, false, false);
                    acc[g][m] = d;
                }
            }
        }

        // ---- LSTM epilogue for this cell (fully wave-local) ----
        const float bi = b_gates[cell * 1024 + 0 * 256 + o];
        const float bj = b_gates[cell * 1024 + 1 * 256 + o];
        const float bf = b_gates[cell * 1024 + 2 * 256 + o];
        const float bo = b_gates[cell * 1024 + 3 * 256 + o];
        #pragma unroll
        for (int m = 0; m < 2; ++m) {
            #pragma unroll
            for (int j = 0; j < 8; ++j) {
                const int lr = m * 16 + j + 8 * laneHi;
                const float gv  = gate_s[lr][cell];
                const float iv  = fast_sigmoid(acc[0][m][j] + bi);
                const float jv  = fast_tanh   (acc[1][m][j] + bj);
                const float fv  = fast_sigmoid(acc[2][m][j] + bf);
                const float ov  = fast_sigmoid(acc[3][m][j] + bo);
                const float ncv = fv * cfrag[m][j] + iv * jv;
                const float nhv = ov * fast_tanh(ncv);
                nc_acc[m][j] += gv * ncv;
                nh_acc[m][j] += gv * nhv;
            }
        }
    }

    // ---- Store: nh first, then nc (tuple return order) ----
    #pragma unroll
    for (int m = 0; m < 2; ++m) {
        #pragma unroll
        for (int j = 0; j < 8; ++j) {
            const int lr = m * 16 + j + 8 * laneHi;
            out[(size_t)(row0 + lr) * OUT_DIM + o] = nh_acc[m][j];
            out[(size_t)B_TOT * OUT_DIM + (size_t)(row0 + lr) * OUT_DIM + o] = nc_acc[m][j];
        }
    }
}

extern "C" void kernel_launch(void* const* d_in, const int* in_sizes, int n_in,
                              void* d_out, int out_size, void* d_ws, size_t ws_size,
                              hipStream_t stream) {
    const float* x       = (const float*)d_in[0];
    const float* c       = (const float*)d_in[1];
    const float* h       = (const float*)d_in[2];
    const float* W_gates = (const float*)d_in[3];
    const float* b_gates = (const float*)d_in[4];
    const float* W_ctrl  = (const float*)d_in[5];
    const float* b_ctrl  = (const float*)d_in[6];
    float* out = (float*)d_out;

    _Float16* Whi = (_Float16*)d_ws;                       // 4096*512 f16 = 4 MB
    _Float16* Wlo = Whi + (size_t)NCOLS * D_DIM;           // 4 MB more

    dim3 tgrid(NCOLS / 32, D_DIM / 32);                    // 128 x 16
    transpose_convert<<<tgrid, 256, 0, stream>>>(W_gates, Whi, Wlo);
    lstm_moe_kernel<<<B_TOT / BM, 512, 0, stream>>>(x, c, h, b_gates,
                                                    W_ctrl, b_ctrl,
                                                    Whi, Wlo, out);
}
